// scalarProductAttentiont_66262755442756
// MI455X (gfx1250) — matrix operand order repaired and checked by reading, not compiled, not run on hardware
//
#include <hip/hip_runtime.h>

// Problem constants (from reference: B=4, N=4096, F=256, fp32 in/out)
#define BATCH 4
#define NN    4096
#define FF    256
#define CH    32      // keys per flash-attention chunk (TDM double-buffered)
#define NCH   (NN / CH)
#define WPB   4       // waves per block (wave32)
#define KPAD  264     // 256 bf16 + 8 bf16 (16B) TDM pad -> 528B row stride

typedef __bf16  bhalf;
typedef __bf16  bhalf8   __attribute__((ext_vector_type(8)));
typedef __bf16  bhalf16  __attribute__((ext_vector_type(16)));
typedef float   vfloat8  __attribute__((ext_vector_type(8)));
typedef unsigned int u32x4 __attribute__((ext_vector_type(4)));
typedef int          i32x8 __attribute__((ext_vector_type(8)));
typedef int          i32x4 __attribute__((ext_vector_type(4)));

// softmax scale folded into exp2: (1/sqrt(F)) * log2(e)
#define KSC 0.09016844005556021f

static __device__ __forceinline__ bhalf16 join16(bhalf8 lo, bhalf8 hi) {
  return __builtin_shufflevector(lo, hi, 0,1,2,3,4,5,6,7,8,9,10,11,12,13,14,15);
}

static __device__ __forceinline__ vfloat8 wmma_bf16(bhalf16 a, bhalf16 b, vfloat8 c) {
  // D = A(16x32 bf16) x B(32x16 bf16) + C(16x16 f32)
  return __builtin_amdgcn_wmma_f32_16x16x32_bf16(false, a, false, b, (short)0, c,
                                                 false, false);
}

// TDM: stage one K chunk (CH rows x 256 bf16, row-major) into LDS with 16B
// per-row padding (row stride 528B). D# per CDNA5 ISA.
// amdgpu-toolchain (clang-23) 6-arg builtin form.
static __device__ __forceinline__ void tdm_load_k(unsigned lds_off,
                                                  unsigned long long ga) {
  u32x4 g0 = {
    1u,                                              // count=1 valid descriptor
    lds_off,                                         // lds_addr (bytes)
    (unsigned)(ga & 0xffffffffull),                  // global_addr[31:0]
    (unsigned)((ga >> 32) & 0x01ffffffull) | 0x80000000u  // addr[56:32]|type=2
  };
  i32x8 g1 = {
    (int)((1u << 16)        // data_size = 2 bytes
        | (1u << 20)        // pad_enable
        | (6u << 22)        // pad_interval: 128 DWORDs (= one 512B row)
        | (3u << 25)),      // pad_amount: 4 DWORDs (16B)
    (int)(256u << 16),      // tensor_dim0 = 256 (bits[79:48], low 16 here)
    (int)(4096u << 16),     // dim0 hi = 0 | tensor_dim1 = 4096 (low 16)
    (int)(256u << 16),      // dim1 hi = 0 | tile_dim0 = 256
    (int)CH,                // tile_dim1 = CH, tile_dim2 = 0
    (int)256,               // tensor_dim0_stride = 256 elements
    0, 0                    // stride hi / tensor_dim1_stride = 0 (2D)
  };
  i32x4 gz  = {0, 0, 0, 0};                  // groups 2/3 unused (2D tensor)
  i32x8 gz8 = {0, 0, 0, 0, 0, 0, 0, 0};      // extra group (disabled)
  __builtin_amdgcn_tensor_load_to_lds(g0, g1, gz, gz, gz8, 0);
}

// ---------------------------------------------------------------------------
// Kernel 0a: Wt[g][k] = bf16(W[k][g])   (256x256, tiny)
// ---------------------------------------------------------------------------
__global__ __launch_bounds__(256) void wt_kernel(const float* __restrict__ W,
                                                 bhalf* __restrict__ Wt) {
  int idx = blockIdx.x * 256 + threadIdx.x;
  int g = idx & 255;
  int k = idx >> 8;
  Wt[g * FF + k] = (bhalf)W[k * FF + g];
}

// ---------------------------------------------------------------------------
// Kernel 0b: nodes(f32) -> nb bf16 [B,N,F] and nt bf16 [B,F,N] (LDS transpose)
// ---------------------------------------------------------------------------
__global__ __launch_bounds__(256) void conv_kernel(const float* __restrict__ nodes,
                                                   bhalf* __restrict__ nb,
                                                   bhalf* __restrict__ nt) {
  __shared__ bhalf tile[32][33];
  int blk = blockIdx.x;                 // BATCH * (NN/32) * (FF/32) = 4096
  int b   = blk >> 10;
  int rem = blk & 1023;
  int nb_i = (rem >> 3) * 32;
  int fb   = (rem & 7) * 32;
  int tx = threadIdx.x & 31;
  int ty = threadIdx.x >> 5;
#pragma unroll
  for (int j = 0; j < 4; ++j) {
    int row = ty + j * 8;
    size_t src = ((size_t)b * NN + nb_i + row) * FF + fb + tx;
    bhalf x = (bhalf)nodes[src];
    nb[src] = x;
    tile[row][tx] = x;
  }
  __syncthreads();
#pragma unroll
  for (int j = 0; j < 4; ++j) {
    int row = ty + j * 8;
    nt[((size_t)b * FF + fb + row) * NN + nb_i + tx] = tile[tx][row];
  }
}

// ---------------------------------------------------------------------------
// Kernel 1: xb = bf16(nodes @ W)  via WMMA. One wave per 16-row tile.
// ---------------------------------------------------------------------------
__global__ __launch_bounds__(32 * WPB) void xw_kernel(const bhalf* __restrict__ nb,
                                                      const bhalf* __restrict__ Wt,
                                                      bhalf* __restrict__ xb) {
  const int lane = threadIdx.x & 31;
  const int w    = threadIdx.x >> 5;
  const int t    = blockIdx.x * WPB + w;
  const int b    = t / (NN / 16);
  const int mb   = (t % (NN / 16)) * 16;
  const int r    = lane & 15;
  const int h    = lane >> 4;

  vfloat8 acc[16];
#pragma unroll
  for (int nt_ = 0; nt_ < 16; ++nt_) acc[nt_] = {};

  const bhalf* arow = nb + ((size_t)b * NN + mb + r) * FF;
#pragma unroll
  for (int kc = 0; kc < 8; ++kc) {
    bhalf8 lo = *(const bhalf8*)(arow + kc * 32 + 8 * h);
    bhalf8 hi = *(const bhalf8*)(arow + kc * 32 + 16 + 8 * h);
    bhalf16 a = join16(lo, hi);
#pragma unroll
    for (int nt_ = 0; nt_ < 16; ++nt_) {
      const bhalf* wp = Wt + (size_t)(nt_ * 16 + r) * FF + kc * 32 + 8 * h;
      bhalf16 bb = join16(*(const bhalf8*)wp, *(const bhalf8*)(wp + 16));
      acc[nt_] = wmma_bf16(a, bb, acc[nt_]);
    }
  }

  bhalf* xrow = xb + ((size_t)b * NN + mb) * FF;
#pragma unroll
  for (int nt_ = 0; nt_ < 16; ++nt_)
#pragma unroll
    for (int d = 0; d < 8; ++d)
      xrow[(size_t)(d + 8 * h) * FF + nt_ * 16 + r] = (bhalf)acc[nt_][d];
}

// ---------------------------------------------------------------------------
// Kernel 2: fused flash attention with TDM-staged K chunks.
// One wave per 16-query tile; K chunks (CH keys) double-buffered in LDS via
// tensor_load_to_lds issued by wave 0; V read from global (L1-shared reuse).
// ---------------------------------------------------------------------------
__global__ __launch_bounds__(32 * WPB) void attn_kernel(const bhalf* __restrict__ xb,
                                                        const bhalf* __restrict__ kb,
                                                        const bhalf* __restrict__ vt,
                                                        float* __restrict__ out) {
  __shared__ __align__(16) bhalf Klds[2][CH][KPAD];      // 2 x 16.5KB
  __shared__ __align__(16) bhalf plds[WPB][16][CH + 8];  // per-wave P staging
  const int lane = threadIdx.x & 31;
  const int w    = threadIdx.x >> 5;
  const int t    = blockIdx.x * WPB + w;
  const int b    = t / (NN / 16);
  const int mb   = (t % (NN / 16)) * 16;
  const int r    = lane & 15;
  const int h    = lane >> 4;

  const bhalf* kbp = kb + (size_t)b * NN * FF;  // [N][F] bf16
  const bhalf* vbp = vt + (size_t)b * FF * NN;  // [F][N] bf16

  unsigned koff0 = (unsigned)(size_t)&Klds[0][0][0];
  unsigned koff1 = (unsigned)(size_t)&Klds[1][0][0];

  // Load the whole 16x256 x-tile as A-fragments (held in registers).
  bhalf16 xa[8];
  const bhalf* xrow = xb + ((size_t)b * NN + mb + r) * FF;
#pragma unroll
  for (int fc = 0; fc < 8; ++fc) {
    bhalf8 lo = *(const bhalf8*)(xrow + fc * 32 + 8 * h);
    bhalf8 hi = *(const bhalf8*)(xrow + fc * 32 + 16 + 8 * h);
    xa[fc] = join16(lo, hi);
  }

  vfloat8 o[16];
#pragma unroll
  for (int ft = 0; ft < 16; ++ft) o[ft] = {};
  float mrow[8], lrow[8];
#pragma unroll
  for (int d = 0; d < 8; ++d) { mrow[d] = -3.0e38f; lrow[d] = 0.0f; }

  // Prologue: stage chunk 0 into buffer 0.
  if (w == 0) tdm_load_k(koff0, (unsigned long long)(const void*)kbp);

  for (int ci = 0; ci < NCH; ++ci) {
    const int k0  = ci * CH;
    const int buf = ci & 1;

    if (w == 0) {
      if (ci + 1 < NCH) {
        tdm_load_k((ci & 1) ? koff0 : koff1,
                   (unsigned long long)(const void*)(kbp + (size_t)(ci + 1) * CH * FF));
        __builtin_amdgcn_s_wait_tensorcnt(1);   // chunk ci landed (in-order)
      } else {
        __builtin_amdgcn_s_wait_tensorcnt(0);
      }
    }
    __syncthreads();  // K chunk ci visible to all waves

    if (ci + 1 < NCH) {  // hint next V stripe into cache
      __builtin_prefetch(vbp + (size_t)(lane * 8) * NN + k0 + CH, 0, 0);
    }

    // ---- S = x_tile @ K_chunk^T from LDS: 2 tiles of 16x16, K-dim = 256 ----
    vfloat8 s[CH / 16];
#pragma unroll
    for (int kt = 0; kt < CH / 16; ++kt) {
      vfloat8 acc = {};
      const bhalf* krow = &Klds[buf][kt * 16 + r][8 * h];
#pragma unroll
      for (int fc = 0; fc < 8; ++fc) {
        bhalf16 bb = join16(*(const bhalf8*)(krow + fc * 32),
                            *(const bhalf8*)(krow + fc * 32 + 16));
        acc = wmma_bf16(xa[fc], bb, acc);
      }
      s[kt] = acc;
    }

    // ---- online softmax; D-layout element = (row d+8h, col kt*16+r) ----
#pragma unroll
    for (int d = 0; d < 8; ++d) {
      float rm = s[0][d];
#pragma unroll
      for (int kt = 1; kt < CH / 16; ++kt) rm = fmaxf(rm, s[kt][d]);
#pragma unroll
      for (int m = 1; m < 16; m <<= 1) rm = fmaxf(rm, __shfl_xor(rm, m));
      float mnew  = fmaxf(mrow[d], rm);
      float alpha = exp2f((mrow[d] - mnew) * KSC);
      mrow[d] = mnew;
      float rs = 0.0f;
#pragma unroll
      for (int kt = 0; kt < CH / 16; ++kt) {
        float p = exp2f((s[kt][d] - mnew) * KSC);
        rs += p;
        plds[w][d + 8 * h][kt * 16 + r] = (bhalf)p;
      }
#pragma unroll
      for (int m = 1; m < 16; m <<= 1) rs += __shfl_xor(rs, m);
      lrow[d] = lrow[d] * alpha + rs;
#pragma unroll
      for (int ft = 0; ft < 16; ++ft) o[ft][d] = o[ft][d] * alpha;
    }

    // ---- O += P @ V_chunk : P from LDS in A-fragment layout, V global ----
    {
      bhalf8 plo = *(const bhalf8*)&plds[w][r][8 * h];
      bhalf8 phi = *(const bhalf8*)&plds[w][r][16 + 8 * h];
      bhalf16 pa = join16(plo, phi);
#pragma unroll
      for (int ft = 0; ft < 16; ++ft) {
        const bhalf* vp = vbp + (size_t)(ft * 16 + r) * NN + k0 + 8 * h;
        bhalf16 vb = join16(*(const bhalf8*)vp, *(const bhalf8*)(vp + 16));
        o[ft] = wmma_bf16(pa, vb, o[ft]);
      }
    }

    __syncthreads();  // all waves done with buffer before TDM reuses it
  }

  // ---- normalize and store fp32 output ----
  float* orow = out + ((size_t)b * NN + mb) * FF;
#pragma unroll
  for (int d = 0; d < 8; ++d) {
    float inv = 1.0f / lrow[d];
#pragma unroll
    for (int ft = 0; ft < 16; ++ft)
      orow[(size_t)(d + 8 * h) * FF + ft * 16 + r] = o[ft][d] * inv;
  }
}

// ---------------------------------------------------------------------------
extern "C" void kernel_launch(void* const* d_in, const int* in_sizes, int n_in,
                              void* d_out, int out_size, void* d_ws, size_t ws_size,
                              hipStream_t stream) {
  (void)in_sizes; (void)n_in; (void)out_size; (void)ws_size;
  const float* nodes = (const float*)d_in[0];   // [B,N,F] f32
  const float* W     = (const float*)d_in[1];   // [F,F]   f32
  float* out = (float*)d_out;                   // [B,N,F] f32

  // Workspace carve-out (bf16): Wt | nodes_b | nodes_t | xb  (~25.3 MB)
  bhalf* Wt = (bhalf*)d_ws;
  bhalf* nb = Wt + (size_t)FF * FF;
  bhalf* nt = nb + (size_t)BATCH * NN * FF;
  bhalf* xb = nt + (size_t)BATCH * NN * FF;

  wt_kernel<<<(FF * FF) / 256, 256, 0, stream>>>(W, Wt);
  conv_kernel<<<BATCH * (NN / 32) * (FF / 32), 256, 0, stream>>>(nodes, nb, nt);
  xw_kernel<<<(BATCH * NN / 16) / WPB, 32 * WPB, 0, stream>>>(nb, Wt, xb);
  attn_kernel<<<(BATCH * NN / 16) / WPB, 32 * WPB, 0, stream>>>(xb, nb, nt, out);
}
